// RPNProposalSSD_52639119180041
// MI455X (gfx1250) — compile-verified, matrix operand order checked by reading, not run on hardware
//
#include <hip/hip_runtime.h>
#include <math.h>

// ---------------------------------------------------------------------------
// RPNProposalSSD for MI455X (gfx1250, wave32).
//   K1: proposal transform + filter          (elementwise, 17340 anchors)
//   K2: stable rank via WMMA ones-matmul     (v_wmma_f32_16x16x32_f16),
//       diagonal-split loop: >=, full tie-break (<=2 iters), >
//   K3: greedy NMS in one WGP's LDS          (120KB dynamic LDS, <=320KB ok)
// ---------------------------------------------------------------------------

typedef __attribute__((ext_vector_type(16))) _Float16 v16h;
typedef __attribute__((ext_vector_type(8)))  float    v8f;

#define A_NUM    15
#define HW_DIM   34
#define SPATIAL  1156        // 34*34
#define N_TOT    17340       // 34*34*15
#define N_PAD    17408       // multiple of 64 (2x unrolled K-tile) and 16
#define N_TILES  (N_PAD / 16)   // 1088
#define MAX_CAND 5000
#define TOP_N    300
#define CAND_PAD 5024

__constant__ float c_AW[15] = {45.3f,32.0f,22.6f,90.5f,64.0f,45.3f,181.0f,128.0f,
                               90.5f,271.5f,192.0f,135.8f,362.0f,256.0f,181.0f};
__constant__ float c_AH[15] = {22.6f,32.0f,45.3f,45.3f,64.0f,90.5f,90.5f,128.0f,
                               181.0f,135.8f,192.0f,271.5f,181.0f,256.0f,362.0f};

// ---------------------------------------------------------------------------
// K1: per-anchor proposal. n = p*15 + a, p = h*34 + w.
// ---------------------------------------------------------------------------
__global__ void __launch_bounds__(256)
k_proposals(const float* __restrict__ cls,
            const float* __restrict__ pred,
            const int*   __restrict__ im_info,
            const float* __restrict__ bmean,
            const float* __restrict__ bstd,
            float*  __restrict__ scoreM,
            float4* __restrict__ boxes,
            int*    __restrict__ cand_valid,
            float*  __restrict__ out)
{
    int n = blockIdx.x * blockDim.x + threadIdx.x;

    if (n < TOP_N * 5)  out[n] = 0.0f;
    if (n < CAND_PAD)   cand_valid[n] = 0;

    if (n >= N_PAD) return;
    if (n >= N_TOT) { scoreM[n] = -1.0e30f; return; }   // pad rows: never rank

    int a = n % A_NUM;
    int p = n / A_NUM;            // == h*34 + w
    int h = p / HW_DIM;
    int w = p % HW_DIM;

    float d0 = pred[(a * 4 + 0) * SPATIAL + p] * bstd[0] + bmean[0];
    float d1 = pred[(a * 4 + 1) * SPATIAL + p] * bstd[1] + bmean[1];
    float d2 = pred[(a * 4 + 2) * SPATIAL + p] * bstd[2] + bmean[2];
    float d3 = pred[(a * 4 + 3) * SPATIAL + p] * bstd[3] + bmean[3];

    float aw = c_AW[a], ah = c_AH[a];
    float ctr_x = (float)w * 16.0f;
    float ctr_y = (float)h * 16.0f;

    float pcx = d0 * aw + ctr_x;
    float pcy = d1 * ah + ctr_y;
    float pw  = __expf(d2) * aw;
    float ph  = __expf(d3) * ah;

    float x1 = pcx - 0.5f * (pw - 1.0f);
    float y1 = pcy - 0.5f * (ph - 1.0f);
    float x2 = pcx + 0.5f * (pw - 1.0f);
    float y2 = pcy + 0.5f * (ph - 1.0f);

    float fw = (float)im_info[1] - 1.0f;
    float fh = (float)im_info[0] - 1.0f;
    x1 = fminf(fmaxf(x1, 0.0f), fw);
    x2 = fminf(fmaxf(x2, 0.0f), fw);
    y1 = fminf(fmaxf(y1, 0.0f), fh);
    y2 = fminf(fmaxf(y2, 0.0f), fh);

    float ws_ = x2 - x1 + 1.0f;
    float hs_ = y2 - y1 + 1.0f;
    float s   = cls[(A_NUM + a) * SPATIAL + p];
    bool keep = (s > 0.2f) && (ws_ >= 6.0f) && (hs_ >= 6.0f);

    scoreM[n] = keep ? s : -1.0f;
    boxes[n]  = make_float4(x1, y1, x2, y2);
}

// ---------------------------------------------------------------------------
// K2: exact stable rank via WMMA. One wave per 16-row tile of rows i.
//   rank(i) = #{ j : s_j > s_i  or  (s_j == s_i and j < i) }
// A-matrix (16x32 f16) ISA layout: lane<16 -> M=lane, K in {0..7,16..23};
// lane>=16 -> M=lane-16, K in {8..15,24..31}; v16h elem e -> K = k0+e(+8, e>=8).
// B = all-ones (value-uniform => layout-irrelevant); C (f32) holds exact counts.
// Diagonal split: tiles fully below the diagonal use (s_j >= s_i), tiles fully
// above use (s_j > s_i) -- 1 cmp + 1 cndmask per element; only the <=2 tiles
// straddling j==i take the 3-compare tie-break (int cmp vs inline literal).
// ---------------------------------------------------------------------------
struct STile { float4 p0, p1, p2, p3; };

__device__ __forceinline__ STile load_tile(const float* __restrict__ sp)
{
    STile t;
    t.p0 = *(const float4*)(sp + 0);    // K = k0+0..3
    t.p1 = *(const float4*)(sp + 4);    // K = k0+4..7
    t.p2 = *(const float4*)(sp + 16);   // K = k0+16..19
    t.p3 = *(const float4*)(sp + 20);   // K = k0+20..23
    return t;
}

// MODE: 0 = (sj > si), 1 = (sj >= si), 2 = full tie-break using relk0
template<int MODE>
__device__ __forceinline__ v16h build_ind(const STile& t, float si, int relk0)
{
    const float sv[16] = {t.p0.x, t.p0.y, t.p0.z, t.p0.w,
                          t.p1.x, t.p1.y, t.p1.z, t.p1.w,
                          t.p2.x, t.p2.y, t.p2.z, t.p2.w,
                          t.p3.x, t.p3.y, t.p3.z, t.p3.w};
    v16h av;
#pragma unroll
    for (int e = 0; e < 16; ++e) {
        const int eoff = e + ((e >= 8) ? 8 : 0);   // K - k0, compile-time literal
        float sj = sv[e];
        int ind;
        if (MODE == 0)      ind = (int)(sj > si);
        else if (MODE == 1) ind = (int)(sj >= si);
        else                ind = (int)(sj > si) |
                                  ((int)(sj == si) & (int)(eoff < relk0));
        av[e] = ind ? (_Float16)1.0f : (_Float16)0.0f;
    }
    return av;
}

__global__ void __launch_bounds__(32)
k_rank(const float*  __restrict__ scoreM,
       const float4* __restrict__ boxes,
       float4* __restrict__ cand_boxes,
       int*    __restrict__ cand_valid)
{
    const int lane  = threadIdx.x;
    const int ibase = blockIdx.x * 16;
    const int ig    = ibase + (lane & 15);     // this lane's row index (M)
    const float si  = scoreM[ig];
    const int k0    = (lane >= 16) ? 8 : 0;

    v16h ones;
#pragma unroll
    for (int e = 0; e < 16; ++e) ones[e] = (_Float16)1.0f;

    v8f c = {0.f, 0.f, 0.f, 0.f, 0.f, 0.f, 0.f, 0.f};

    // wave-uniform range bounds (ibase is block-uniform, multiple of 16)
    const int r1_end = ibase & ~63;              // jb < r1_end  => j <  ig always
    const int r2_end = (ibase + 79) & ~63;       // jb >= r2_end => j >  ig always
    int jb = 0;

    // --- below diagonal: ties count (j < i) -> sj >= si -------------------
    for (; jb < r1_end; jb += 64) {
        const float* sp = scoreM + jb + k0;
        __builtin_prefetch(sp + 512, 0, 1);
        STile t0 = load_tile(sp);
        STile t1 = load_tile(sp + 32);
        v16h a0 = build_ind<1>(t0, si, 0);
        v16h a1 = build_ind<1>(t1, si, 0);
        c = __builtin_amdgcn_wmma_f32_16x16x32_f16(false, a0, false, ones,
                                                   (short)0, c, false, false);
        c = __builtin_amdgcn_wmma_f32_16x16x32_f16(false, a1, false, ones,
                                                   (short)0, c, false, false);
    }
    // --- diagonal band (<= 2 iterations): full tie-break -------------------
    for (; jb < r2_end; jb += 64) {
        const float* sp = scoreM + jb + k0;
        STile t0 = load_tile(sp);
        STile t1 = load_tile(sp + 32);
        v16h a0 = build_ind<2>(t0, si, ig - jb - k0);
        v16h a1 = build_ind<2>(t1, si, ig - (jb + 32) - k0);
        c = __builtin_amdgcn_wmma_f32_16x16x32_f16(false, a0, false, ones,
                                                   (short)0, c, false, false);
        c = __builtin_amdgcn_wmma_f32_16x16x32_f16(false, a1, false, ones,
                                                   (short)0, c, false, false);
    }
    // --- above diagonal: ties don't count (j > i) -> sj > si ---------------
    for (; jb < N_PAD; jb += 64) {
        const float* sp = scoreM + jb + k0;
        __builtin_prefetch(sp + 512, 0, 1);
        STile t0 = load_tile(sp);
        STile t1 = load_tile(sp + 32);
        v16h a0 = build_ind<0>(t0, si, 0);
        v16h a1 = build_ind<0>(t1, si, 0);
        c = __builtin_amdgcn_wmma_f32_16x16x32_f16(false, a0, false, ones,
                                                   (short)0, c, false, false);
        c = __builtin_amdgcn_wmma_f32_16x16x32_f16(false, a1, false, ones,
                                                   (short)0, c, false, false);
    }

    // C/D layout: VGPR r holds M=r (lanes 0-15) / M=r+8 (lanes 16-31); all N equal.
    if (lane == 0 || lane == 16) {
        int mb = (lane == 16) ? 8 : 0;
#pragma unroll
        for (int r = 0; r < 8; ++r) {
            int i = ibase + mb + r;
            if (i < N_TOT) {
                int rk = (int)(c[r] + 0.5f);
                if (rk < MAX_CAND) {
                    cand_boxes[rk] = boxes[i];
                    cand_valid[rk] = (scoreM[i] > 0.0f) ? 1 : 0;
                }
            }
        }
    }
}

// ---------------------------------------------------------------------------
// K3: greedy NMS, single workgroup, everything in LDS (6*5024*4 = 120576 B,
// fine under CDNA5's 320KB/WGP). One barrier per candidate; each j owned by
// exactly one thread per iteration -> race-free plain LDS stores.
// ---------------------------------------------------------------------------
__global__ void __launch_bounds__(1024)
k_nms(const float4* __restrict__ cand_boxes,
      const int*    __restrict__ cand_valid,
      float* __restrict__ out)
{
    extern __shared__ float smem[];
    float* X1 = smem;
    float* Y1 = X1 + CAND_PAD;
    float* X2 = Y1 + CAND_PAD;
    float* Y2 = X2 + CAND_PAD;
    float* AR = Y2 + CAND_PAD;
    int*   SUP = (int*)(AR + CAND_PAD);

    const int t = threadIdx.x;
    for (int m = t; m < MAX_CAND; m += 1024) {
        float4 b = cand_boxes[m];
        X1[m] = b.x; Y1[m] = b.y; X2[m] = b.z; Y2[m] = b.w;
        AR[m] = (b.z - b.x + 1.0f) * (b.w - b.y + 1.0f);
        SUP[m] = cand_valid[m] ? 0 : 1;   // invalid -> pre-suppressed
    }

    for (int i = 0; i < MAX_CAND; ++i) {
        __syncthreads();                   // publish prior iteration's SUP writes
        if (SUP[i]) continue;              // uniform value -> uniform branch
        float x1i = X1[i], y1i = Y1[i], x2i = X2[i], y2i = Y2[i], ari = AR[i];
        for (int j = i + 1 + t; j < MAX_CAND; j += 1024) {
            if (!SUP[j]) {
                float iw = fminf(x2i, X2[j]) - fmaxf(x1i, X1[j]) + 1.0f;
                float ih = fminf(y2i, Y2[j]) - fmaxf(y1i, Y1[j]) + 1.0f;
                iw = fmaxf(iw, 0.0f);
                ih = fmaxf(ih, 0.0f);
                float inter = iw * ih;
                float iou = inter / (ari + AR[j] - inter);
                if (iou > 0.7f) SUP[j] = 1;
            }
        }
    }
    __syncthreads();

    if (t == 0) {                          // compaction: trivial vs. NMS chain
        int cnt = 0;
        for (int j = 0; j < MAX_CAND && cnt < TOP_N; ++j) {
            if (!SUP[j]) {
                out[cnt * 5 + 1] = X1[j];
                out[cnt * 5 + 2] = Y1[j];
                out[cnt * 5 + 3] = X2[j];
                out[cnt * 5 + 4] = Y2[j];
                ++cnt;                     // col 0 / tail rows stay 0 from K1
            }
        }
    }
}

// ---------------------------------------------------------------------------
extern "C" void kernel_launch(void* const* d_in, const int* in_sizes, int n_in,
                              void* d_out, int out_size, void* d_ws, size_t ws_size,
                              hipStream_t stream)
{
    const float* cls    = (const float*)d_in[0];   // (1,30,34,34)
    const float* pred   = (const float*)d_in[1];   // (1,60,34,34)
    const int*   iminfo = (const int*)  d_in[2];   // [H,W] = [544,544]
    const float* bmean  = (const float*)d_in[3];
    const float* bstd   = (const float*)d_in[4];
    float* out = (float*)d_out;                    // (300,5) f32

    // workspace layout (all 16B aligned); N_PAD*4 = 69632
    char* ws = (char*)d_ws;
    float*  scoreM     = (float*) (ws);                              // N_PAD f32
    float4* boxes      = (float4*)(ws + 69632);                      // N_TOT f4 (277440 B)
    float4* cand_boxes = (float4*)(ws + 69632 + 277440);             // 5000 f4  (80000 B)
    int*    cand_valid = (int*)   (ws + 69632 + 277440 + 80000);     // CAND_PAD i32

    k_proposals<<<(N_PAD + 255) / 256, 256, 0, stream>>>(
        cls, pred, iminfo, bmean, bstd, scoreM, boxes, cand_valid, out);

    k_rank<<<N_TILES, 32, 0, stream>>>(scoreM, boxes, cand_boxes, cand_valid);

    k_nms<<<1, 1024, 6 * CAND_PAD * sizeof(float), stream>>>(
        cand_boxes, cand_valid, out);
}
